// FCUDown_5961414607113
// MI455X (gfx1250) — matrix operand order repaired and checked by reading, not compiled, not run on hardware
//
#include <hip/hip_runtime.h>
#include <hip/hip_bf16.h>
#include <math.h>

typedef __attribute__((ext_vector_type(16))) _Float16 v16h;
typedef __attribute__((ext_vector_type(8)))  float    v8f;
typedef __attribute__((ext_vector_type(4)))  float    v4f;

#define CIN   1536
#define COUT  3072
#define NT    16
#define NPIX  196          // 14*14
#define NPATCH 49          // 7*7
#define NP    784          // 16*49
#define YS_T_BYTES (NP * CIN * 2)   // 2,408,448

// ---------------------------------------------------------------------------
// Kernel 1: gather strided slice y[:, :, t, ::2, ::2] -> f16 ysT[n=784][c=1536]
// LDS transpose so global writes are contiguous in c.
// grid = (CIN/64, 16), block = 256
// ---------------------------------------------------------------------------
__global__ void build_ysT(const float* __restrict__ y, _Float16* __restrict__ ysT) {
    __shared__ _Float16 sh[64 * NPATCH];
    const int t  = blockIdx.y;
    const int c0 = blockIdx.x * 64;
    for (int idx = threadIdx.x; idx < 64 * NPATCH; idx += 256) {
        const int cl = idx / NPATCH;
        const int nl = idx % NPATCH;
        const int h  = nl / 7, w = nl % 7;
        sh[cl * NPATCH + nl] =
            (_Float16)y[(size_t)(c0 + cl) * (NT * NPIX) + t * NPIX + (2 * h) * 14 + 2 * w];
    }
    __syncthreads();
    for (int idx = threadIdx.x; idx < 64 * NPATCH; idx += 256) {
        const int nl = idx / 64;
        const int cl = idx % 64;
        ysT[(size_t)(t * NPATCH + nl) * CIN + c0 + cl] = sh[cl * NPATCH + nl];
    }
}

// ---------------------------------------------------------------------------
// Kernel 2: yp[o][n] = GELU(BN(conv_w x ysT + b)) via v_wmma_f32_16x16x32_f16.
// 8 waves/WG; each wave computes a 16(M) x 112(N = 7 tiles) strip so one
// A fragment (f32->f16 converted on the fly) feeds 7 WMMAs per K step.
//   A 16x32 f16 layout: lane (g=l/16, m=l%16); 16 contiguous halves at
//     K kb+g*16..+15 (four v4f loads, cvt to f16)
//   B 32x16 f16 layout: lane (g, n=l%16); 16 contiguous halves at k=kb+g*16
//   C/D: lane half g: elem j -> M = j + 8*g, N = l%16
// grid = (COUT/16/8 = 24, 49/7 = 7), block = 256
// ---------------------------------------------------------------------------
__global__ void gemm_bn_gelu(const float*    __restrict__ conv_w,
                             const _Float16* __restrict__ ysT,
                             const float* __restrict__ conv_b,
                             const float* __restrict__ gamma,
                             const float* __restrict__ beta,
                             const float* __restrict__ run_mean,
                             const float* __restrict__ run_var,
                             float* __restrict__ yp) {
    const int lane   = threadIdx.x & 31;
    const int wave   = threadIdx.x >> 5;
    const int m_tile = blockIdx.x * 8 + wave;   // 0..191
    const int nt0    = blockIdx.y * 7;          // first of 7 N-tiles
    const int g  = lane >> 4;
    const int mn = lane & 15;

    const float*    aptr = conv_w + (size_t)(m_tile * 16 + mn) * CIN;
    const _Float16* bptr = ysT    + (size_t)(nt0 * 16 + mn) * CIN;   // +i*16*CIN per tile

    v8f c0 = {}, c1 = {}, c2 = {}, c3 = {}, c4 = {}, c5 = {}, c6 = {};

    for (int kb = 0; kb < CIN; kb += 32) {
        // A fragment: 16 f32 -> f16
        const v4f a0 = *(const v4f*)(aptr + kb + g * 16);
        const v4f a1 = *(const v4f*)(aptr + kb + g * 16 + 4);
        const v4f a2 = *(const v4f*)(aptr + kb + g * 16 + 8);
        const v4f a3 = *(const v4f*)(aptr + kb + g * 16 + 12);
        v16h a;
        a[0]  = (_Float16)a0[0]; a[1]  = (_Float16)a0[1];
        a[2]  = (_Float16)a0[2]; a[3]  = (_Float16)a0[3];
        a[4]  = (_Float16)a1[0]; a[5]  = (_Float16)a1[1];
        a[6]  = (_Float16)a1[2]; a[7]  = (_Float16)a1[3];
        a[8]  = (_Float16)a2[0]; a[9]  = (_Float16)a2[1];
        a[10] = (_Float16)a2[2]; a[11] = (_Float16)a2[3];
        a[12] = (_Float16)a3[0]; a[13] = (_Float16)a3[1];
        a[14] = (_Float16)a3[2]; a[15] = (_Float16)a3[3];

        const _Float16* bk = bptr + kb + g * 16;
        v16h b;
        b = *(const v16h*)(bk + 0 * 16 * CIN);
        c0 = __builtin_amdgcn_wmma_f32_16x16x32_f16(false, a, false, b, (short)0, c0, false, false);
        b = *(const v16h*)(bk + 1 * 16 * CIN);
        c1 = __builtin_amdgcn_wmma_f32_16x16x32_f16(false, a, false, b, (short)0, c1, false, false);
        b = *(const v16h*)(bk + 2 * 16 * CIN);
        c2 = __builtin_amdgcn_wmma_f32_16x16x32_f16(false, a, false, b, (short)0, c2, false, false);
        b = *(const v16h*)(bk + 3 * 16 * CIN);
        c3 = __builtin_amdgcn_wmma_f32_16x16x32_f16(false, a, false, b, (short)0, c3, false, false);
        b = *(const v16h*)(bk + 4 * 16 * CIN);
        c4 = __builtin_amdgcn_wmma_f32_16x16x32_f16(false, a, false, b, (short)0, c4, false, false);
        b = *(const v16h*)(bk + 5 * 16 * CIN);
        c5 = __builtin_amdgcn_wmma_f32_16x16x32_f16(false, a, false, b, (short)0, c5, false, false);
        b = *(const v16h*)(bk + 6 * 16 * CIN);
        c6 = __builtin_amdgcn_wmma_f32_16x16x32_f16(false, a, false, b, (short)0, c6, false, false);
    }

    // epilogue: bias + BN + exact GELU; params loaded once per output row
    const int nbase = nt0 * 16 + mn;
#pragma unroll
    for (int j = 0; j < 8; ++j) {
        const int o = m_tile * 16 + j + 8 * g;
        const float bo   = conv_b[o];
        const float inv  = gamma[o] / sqrtf(run_var[o] + 1e-6f);
        const float mu   = run_mean[o];
        const float bet  = beta[o];
        float* yo = yp + (size_t)o * NP + nbase;
        float vv[7] = { c0[j], c1[j], c2[j], c3[j], c4[j], c5[j], c6[j] };
#pragma unroll
        for (int i = 0; i < 7; ++i) {
            float v = vv[i] + bo;
            v = (v - mu) * inv + bet;
            v = 0.5f * v * (1.0f + erff(v * 0.70710678118654752f)); // exact GELU
            yo[i * 16] = v;
        }
    }
}

// ---------------------------------------------------------------------------
// Kernel 3: per (o,t) plane: ROI-align mr, fuse yp*mr scatter + residual add.
// grid = COUT*NT = 49152 blocks of 256 threads (196 pixels active).
// ---------------------------------------------------------------------------
__global__ void roi_combine(const float* __restrict__ x_t,
                            const float* __restrict__ rois,
                            const float* __restrict__ yp,
                            float* __restrict__ out) {
    __shared__ float tile[NPIX];
    const int blk = blockIdx.x;
    const int t = blk & 15;
    const int o = blk >> 4;
    const size_t plane = ((size_t)o * NT + t) * NPIX;
    const int tid = threadIdx.x;
    if (tid < NPIX) tile[tid] = x_t[plane + tid];
    __syncthreads();

    // box parameters (uniform per block)
    const float b1x = rois[t * 4 + 0] * 0.0625f;
    const float b1y = rois[t * 4 + 1] * 0.0625f;
    const float b2x = rois[t * 4 + 2] * 0.0625f;
    const float b2y = rois[t * 4 + 3] * 0.0625f;
    const float rw = fmaxf(b2x - b1x, 1.0f);
    const float rh = fmaxf(b2y - b1y, 1.0f);
    const float bw = rw * (1.0f / 7.0f);
    const float bh = rh * (1.0f / 7.0f);
    const int sxi = (int)floorf(b1x), syi = (int)floorf(b1y);
    const int exi = (int)ceilf(b2x),  eyi = (int)ceilf(b2y);
    int ml = (sxi + 7 < 14) ? sxi : (exi - 7);
    int mt = (syi + 7 < 14) ? syi : (eyi - 7);
    ml = min(max(ml, 0), 7);   // dynamic_update_slice clamping
    mt = min(max(mt, 0), 7);

    if (tid < NPIX) {
        const int yy = tid / 14, xx = tid % 14;
        float v = tile[tid];
        const int ph = yy - mt, pw = xx - ml;
        if (ph >= 0 && ph < 7 && pw >= 0 && pw < 7) {
            float mr = 0.0f;
#pragma unroll
            for (int sy = 0; sy < 2; ++sy)
#pragma unroll
            for (int sx = 0; sx < 2; ++sx) {
                const float cy = b1y + ((float)ph + (sy + 0.5f) * 0.5f) * bh;
                const float cx = b1x + ((float)pw + (sx + 0.5f) * 0.5f) * bw;
                const bool ok = !(cy < -1.0f || cy > 14.0f || cx < -1.0f || cx > 14.0f);
                const float yc = fminf(fmaxf(cy, 0.0f), 13.0f);
                const float xc = fminf(fmaxf(cx, 0.0f), 13.0f);
                const int y0 = (int)floorf(yc), x0 = (int)floorf(xc);
                const int y1 = min(y0 + 1, 13), x1 = min(x0 + 1, 13);
                const float ly = yc - (float)y0, lx = xc - (float)x0;
                const float s =
                    (1.0f - ly) * ((1.0f - lx) * tile[y0 * 14 + x0] + lx * tile[y0 * 14 + x1]) +
                            ly  * ((1.0f - lx) * tile[y1 * 14 + x0] + lx * tile[y1 * 14 + x1]);
                mr += ok ? s : 0.0f;
            }
            mr *= 0.25f;
            v += yp[(size_t)o * NP + t * NPATCH + ph * 7 + pw] * mr;
        }
        out[plane + tid] = v;
    }
}

// ---------------------------------------------------------------------------
extern "C" void kernel_launch(void* const* d_in, const int* in_sizes, int n_in,
                              void* d_out, int out_size, void* d_ws, size_t ws_size,
                              hipStream_t stream) {
    const float* y        = (const float*)d_in[0];  // (1,1536,16,14,14)
    const float* x_t      = (const float*)d_in[1];  // (1,3072,16,14,14)
    const float* rois     = (const float*)d_in[2];  // (16,1,4)
    const float* conv_w   = (const float*)d_in[3];  // (3072,1536)
    const float* conv_b   = (const float*)d_in[4];
    const float* gamma    = (const float*)d_in[5];
    const float* beta     = (const float*)d_in[6];
    const float* run_mean = (const float*)d_in[7];
    const float* run_var  = (const float*)d_in[8];
    float* out = (float*)d_out;

    _Float16* ysT = (_Float16*)d_ws;                         // 2.41 MB
    float*    yp  = (float*)((char*)d_ws + YS_T_BYTES);      // 9.63 MB

    build_ysT<<<dim3(CIN / 64, NT), 256, 0, stream>>>(y, ysT);
    gemm_bn_gelu<<<dim3(COUT / 128, 7), 256, 0, stream>>>(
        conv_w, ysT, conv_b, gamma, beta, run_mean, run_var, yp);
    roi_combine<<<dim3(COUT * NT), 256, 0, stream>>>(x_t, rois, yp, out);
}
